// GraftNet_91053306675397
// MI455X (gfx1250) — matrix operand order repaired and checked by reading, not compiled, not run on hardware
//
#include <hip/hip_runtime.h>
#include <hip/hip_bf16.h>
#include <math.h>

// ---------------- constants (shapes fixed by setup_inputs) ----------------
#define B_   8
#define E_   2000
#define F_   12000
#define Q_   20
#define WD_  300
#define D_   100
#define NW_  40000
#define NE_  100000
#define LAYERS_ 3
#define VERY_NEG  (-100000000000.0f)
#define VERY_SMALL 1e-10f
#define PR_LAMBDA 0.8f
#define FACT_SCALE 3.0f

typedef __attribute__((ext_vector_type(16))) _Float16 v16h;
typedef __attribute__((ext_vector_type(8)))  float    v8f;

__device__ __forceinline__ float sigm(float x) { return 1.0f / (1.0f + __expf(-x)); }

// ---------------- templated WMMA GEMM: C = act(A @ W^T + bias) ----------------
// K compile-time so the unrolled tile loop has constant-foldable bounds:
// full tiles -> unconditional contiguous loads; tail tile -> clamped-index loads
// + value selects (no exec predication anywhere).
//
// Fragment layouts (CDNA5 ISA §7.12.2, wave32):
//   A 16x32 f16 : lane L holds row m=L&15, kh=L>>4; elem e -> K = k0+16*kh+e (e<8)
//                 or k0+16*kh+8+(e-8). Two contiguous 8-float runs.
//   B 32x16 f16 : lane holds col n=L&15, kh=L>>4; elem e -> K = k0+16*kh+e.
//                 B[k][n] = W[n][k], W row-major (N,K) => 16 contiguous floats.
//   C 16x16 f32 : 8 VGPRs; row = r + 8*(L>>4), col = L&15.
// Out-of-range output columns are computed from a clamped weight row and simply
// not stored (WMMA columns are independent).
template <int K>
__global__ void wmma_gemm_kernel(const float* __restrict__ A,
                                 const int* __restrict__ rowidx,
                                 const float* __restrict__ table,
                                 const float* __restrict__ W,
                                 const float* __restrict__ bias,
                                 float* __restrict__ C,
                                 int M, int relu) {
  int mt = blockIdx.x * 4 + (threadIdx.x >> 5);      // wave-uniform
  int m0 = mt * 16;
  if (m0 >= M) return;                               // uniform exit: EXEC all-1 at WMMA
  int n0 = blockIdx.y * 16;
  int lane = threadIdx.x & 31;
  int half = lane >> 4;
  int lid  = lane & 15;

  int arow = m0 + lid;                               // M is 16-aligned for all uses
  const float* rp = rowidx ? (table + (size_t)rowidx[arow] * (size_t)K)
                           : (A + (size_t)arow * (size_t)K);
  int ncol = n0 + lid;
  const float* wp = W + (size_t)(ncol < D_ ? ncol : (D_ - 1)) * (size_t)K;

  v8f acc = {};
#pragma unroll
  for (int k0 = 0; k0 < K; k0 += 32) {
    v16h af, bf;
    if (k0 + 32 <= K) {
      // fast path: all in-bounds, unconditional contiguous loads
      const float* pa1 = rp + k0 + 16 * half;
      const float* pa2 = rp + k0 + 16 * half + 8;
#pragma unroll
      for (int e = 0; e < 8; ++e) { af[e] = (_Float16)pa1[e]; af[8 + e] = (_Float16)pa2[e]; }
      const float* pb = wp + k0 + 16 * half;
#pragma unroll
      for (int e = 0; e < 16; ++e) bf[e] = (_Float16)pb[e];
    } else {
      // tail tile: clamp index for the load, select value (no predicated loads)
      int kb1 = k0 + 16 * half;
      int kb2 = k0 + 16 * half + 8;
#pragma unroll
      for (int e = 0; e < 8; ++e) {
        int k1 = kb1 + e; bool v1 = (k1 < K);
        float x1 = rp[v1 ? k1 : 0];
        af[e] = v1 ? (_Float16)x1 : (_Float16)0.0f;
        int k2 = kb2 + e; bool v2 = (k2 < K);
        float x2 = rp[v2 ? k2 : 0];
        af[8 + e] = v2 ? (_Float16)x2 : (_Float16)0.0f;
      }
      int kb = k0 + 16 * half;
#pragma unroll
      for (int e = 0; e < 16; ++e) {
        int k = kb + e; bool v = (k < K);
        float x = wp[v ? k : 0];
        bf[e] = v ? (_Float16)x : (_Float16)0.0f;
      }
    }
    acc = __builtin_amdgcn_wmma_f32_16x16x32_f16(false, af, false, bf,
                                                 (short)0, acc, false, false);
  }

  if (ncol < D_) {
    float bv = bias[ncol];
#pragma unroll
    for (int r = 0; r < 8; ++r) {
      int row = m0 + r + 8 * half;
      float v = acc[r] + bv;
      if (relu) v = fmaxf(v, 0.0f);
      C[(size_t)row * D_ + ncol] = v;
    }
  }
}

// ---------------- LSTM over query (B blocks, sequential T=20) ----------------
__global__ void lstm_kernel(const float* __restrict__ word_emb,
                            const int* __restrict__ query_text,
                            const float* __restrict__ Wih, const float* __restrict__ Whh,
                            const float* __restrict__ bih, const float* __restrict__ bhh,
                            float* __restrict__ qh, float* __restrict__ q_last) {
  __shared__ float xs[WD_], hs[D_], cs[D_], gs[4 * D_];
  int b = blockIdx.x, tid = threadIdx.x, nt = blockDim.x;
  for (int i = tid; i < D_; i += nt) { hs[i] = 0.0f; cs[i] = 0.0f; }
  __syncthreads();
  for (int t = 0; t < Q_; ++t) {
    int w = query_text[b * Q_ + t];
    for (int i = tid; i < WD_; i += nt) xs[i] = word_emb[(size_t)w * WD_ + i];
    __syncthreads();
    for (int j = tid; j < 4 * D_; j += nt) {
      float g = bih[j] + bhh[j];
      const float* wi = Wih + (size_t)j * WD_;
      for (int k = 0; k < WD_; ++k) g += xs[k] * wi[k];
      const float* wh = Whh + (size_t)j * D_;
      for (int k = 0; k < D_; ++k) g += hs[k] * wh[k];
      gs[j] = g;
    }
    __syncthreads();
    for (int d = tid; d < D_; d += nt) {
      float i_ = sigm(gs[d]);
      float f_ = sigm(gs[D_ + d]);
      float gg = tanhf(gs[2 * D_ + d]);
      float o_ = sigm(gs[3 * D_ + d]);
      float cc = f_ * cs[d] + i_ * gg;
      float hh = o_ * tanhf(cc);
      cs[d] = cc; hs[d] = hh;
      qh[((size_t)b * Q_ + t) * D_ + d] = hh;
      if (t == Q_ - 1) q_last[b * D_ + d] = hh;
    }
    __syncthreads();
  }
}

// ---------------- attention -> Wf (one wave per fact) ----------------
// Wf[b,f] = sum_q softmax_q(s + mask)[q] * s[q],  s[q] = <qh[b,q], lfe[b,f]>/10
__global__ void attn_wf_kernel(const float* __restrict__ qh,
                               const float* __restrict__ lfe,
                               const int* __restrict__ query_text,
                               float* __restrict__ Wf) {
  int wid = blockIdx.x * (blockDim.x >> 5) + (threadIdx.x >> 5);  // fact over B*F
  int lane = threadIdx.x & 31;
  int b = wid / F_;
  const float* fe = lfe + (size_t)wid * D_;
  float f0 = fe[lane], f1 = fe[32 + lane], f2 = fe[64 + lane];
  float f3 = (lane < 4) ? fe[96 + lane] : 0.0f;

  float s[Q_], sm[Q_];
  float mx = -3.4e38f;
#pragma unroll
  for (int q = 0; q < Q_; ++q) {
    const float* qv = qh + ((size_t)b * Q_ + q) * D_;
    float t3 = (lane < 4) ? qv[96 + lane] : 0.0f;
    float p = f0 * qv[lane] + f1 * qv[32 + lane] + f2 * qv[64 + lane] + f3 * t3;
    for (int off = 16; off; off >>= 1) p += __shfl_xor(p, off, 32);
    p *= 0.1f;  // 1/sqrt(D)
    s[q] = p;
    float m = (query_text[b * Q_ + q] != NW_) ? 0.0f : VERY_NEG;
    sm[q] = p + m;
    mx = fmaxf(mx, sm[q]);
  }
  float den = 0.0f;
#pragma unroll
  for (int q = 0; q < Q_; ++q) { sm[q] = __expf(sm[q] - mx); den += sm[q]; }
  float wf = 0.0f;
#pragma unroll
  for (int q = 0; q < Q_; ++q) wf += (sm[q] / den) * s[q];
  if (lane == 0) Wf[wid] = wf;
}

// ---------------- misc elementwise / reduce / scatter kernels ----------------
__global__ void fill_kernel(float* p, int n, float v) {
  int t = blockIdx.x * blockDim.x + threadIdx.x;
  if (t < n) p[t] = v;
}
__global__ void copy_kernel(float* dst, const float* src, int n) {
  int t = blockIdx.x * blockDim.x + threadIdx.x;
  if (t < n) dst[t] = src[t];
}
__global__ void rowmax_kernel(const float* __restrict__ Wf, float* __restrict__ wmax) {
  __shared__ float red[32];
  int b = blockIdx.x, tid = threadIdx.x;
  float m = -3.4e38f;
  for (int f = tid; f < F_; f += blockDim.x) m = fmaxf(m, Wf[b * F_ + f]);
  for (int off = 16; off; off >>= 1) m = fmaxf(m, __shfl_xor(m, off, 32));
  if ((tid & 31) == 0) red[tid >> 5] = m;
  __syncthreads();
  if (tid < 32) {
    m = (tid < (blockDim.x >> 5)) ? red[tid] : -3.4e38f;
    for (int off = 16; off; off >>= 1) m = fmaxf(m, __shfl_xor(m, off, 32));
    if (tid == 0) wmax[b] = m;
  }
}
__global__ void wtilde_kernel(const float* __restrict__ Wf, const float* __restrict__ wmax,
                              const int* __restrict__ e2f, float* __restrict__ Wt,
                              float* __restrict__ e2fsm) {
  int t = blockIdx.x * blockDim.x + threadIdx.x;
  if (t >= B_ * F_) return;
  int b = t / F_;
  float wt = __expf(Wf[t] - wmax[b]);
  Wt[t] = wt;
  atomicAdd(&e2fsm[b * E_ + e2f[t]], wt);
}
__global__ void e2f_norm_kernel(const float* __restrict__ Wt, const float* __restrict__ pr,
                                const float* __restrict__ e2fsm, const int* __restrict__ e2f,
                                float* __restrict__ norm) {
  int t = blockIdx.x * blockDim.x + threadIdx.x;
  if (t >= B_ * F_) return;
  int b = t / F_;
  int ent = e2f[t];
  float sm = fmaxf(e2fsm[b * E_ + ent], VERY_SMALL);
  norm[t] = Wt[t] * (pr[b * E_ + ent] / sm);
}
// in-place: fs[t] = relu(fs[t] + head[b, e2f_entity, d]) * norm
__global__ void e2f_emb_kernel(float* __restrict__ fs, const float* __restrict__ head,
                               const int* __restrict__ e2f, const float* __restrict__ norm) {
  int t = blockIdx.x * blockDim.x + threadIdx.x;
  if (t >= B_ * F_ * D_) return;
  int d = t % D_;
  int bf = t / D_;
  int b = bf / F_;
  int ent = e2f[bf];
  float v = fmaxf(fs[t] + head[((size_t)b * E_ + ent) * D_ + d], 0.0f);
  fs[t] = v * norm[bf];
}
__global__ void scatter_tail_kernel(const float* __restrict__ tail, const int* __restrict__ f2e,
                                    float* __restrict__ acc) {
  int t = blockIdx.x * blockDim.x + threadIdx.x;
  if (t >= B_ * F_ * D_) return;
  int d = t % D_;
  int bf = t / D_;
  int b = bf / F_;
  atomicAdd(&acc[((size_t)b * E_ + f2e[bf]) * D_ + d], tail[t]);
}
__global__ void scatter_pr_kernel(const float* __restrict__ norm, const int* __restrict__ f2e,
                                  float* __restrict__ pracc) {
  int t = blockIdx.x * blockDim.x + threadIdx.x;
  if (t >= B_ * F_) return;
  int b = t / F_;
  atomicAdd(&pracc[b * E_ + f2e[t]], norm[t]);
}
__global__ void pr_update_kernel(float* __restrict__ pr, const float* __restrict__ pracc) {
  int t = blockIdx.x * blockDim.x + threadIdx.x;
  if (t >= B_ * E_) return;
  pr[t] = PR_LAMBDA * pracc[t] + (1.0f - PR_LAMBDA) * pr[t];
}
__global__ void q2e_kernel(const float* __restrict__ qlast, const float* __restrict__ W,
                           const float* __restrict__ bias, float* __restrict__ out) {
  int b = blockIdx.x, d = threadIdx.x;
  if (d < D_) {
    float s = bias[d];
    const float* wr = W + (size_t)d * D_;
    const float* q = qlast + b * D_;
    for (int k = 0; k < D_; ++k) s += q[k] * wr[k];
    out[b * D_ + d] = s;
  }
}
// nxt = concat(lee, q2e_vec broadcast, FACT_SCALE*relu(es+acc)) : (B*E, 300)
__global__ void concat_kernel(const float* __restrict__ lee, const float* __restrict__ qvec,
                              const float* __restrict__ es, const float* __restrict__ acc,
                              float* __restrict__ nxt) {
  int t = blockIdx.x * blockDim.x + threadIdx.x;
  if (t >= B_ * E_ * D_) return;
  int d = t % D_;
  int row = t / D_;
  int b = row / E_;
  size_t base = (size_t)row * (3 * D_);
  nxt[base + d]          = lee[t];
  nxt[base + D_ + d]     = qvec[b * D_ + d];
  nxt[base + 2 * D_ + d] = FACT_SCALE * fmaxf(es[t] + acc[t], 0.0f);
}
__global__ void score_kernel(const float* __restrict__ lee, const float* __restrict__ sw,
                             const float* __restrict__ sb, const int* __restrict__ local_entity,
                             float* __restrict__ out) {
  int t = blockIdx.x * blockDim.x + threadIdx.x;
  if (t >= B_ * E_) return;
  const float* r = lee + (size_t)t * D_;
  float s = sb[0];
  for (int d = 0; d < D_; ++d) s += r[d] * sw[d];
  float mask = (local_entity[t] != NE_) ? 1.0f : 0.0f;
  s += (1.0f - mask) * VERY_NEG;
  out[t] = sigm(s) * mask;
}

// ---------------- host-side orchestration ----------------
static inline size_t align_up(size_t x) { return (x + 255) & ~(size_t)255; }

extern "C" void kernel_launch(void* const* d_in, const int* in_sizes, int n_in,
                              void* d_out, int out_size, void* d_ws, size_t ws_size,
                              hipStream_t stream) {
  (void)in_sizes; (void)n_in; (void)out_size; (void)ws_size;
  const int*   local_entity = (const int*)  d_in[0];
  const float* q2e_adj      = (const float*)d_in[1];
  const int*   kb_fact_rel  = (const int*)  d_in[2];
  const int*   query_text   = (const int*)  d_in[3];
  const int*   e2f_entity   = (const int*)  d_in[4];
  const int*   f2e_entity   = (const int*)  d_in[5];
  const float* word_emb     = (const float*)d_in[7];
  const float* entity_emb   = (const float*)d_in[8];
  const float* rel_emb      = (const float*)d_in[9];
  const float* ent_W  = (const float*)d_in[10];
  const float* ent_b  = (const float*)d_in[11];
  const float* rel_W  = (const float*)d_in[12];
  const float* rel_b  = (const float*)d_in[13];
  const float* lstm_Wih = (const float*)d_in[14];
  const float* lstm_Whh = (const float*)d_in[15];
  const float* lstm_bih = (const float*)d_in[16];
  const float* lstm_bhh = (const float*)d_in[17];
  const float* q2e_W  = (const float*)d_in[18];
  const float* q2e_b  = (const float*)d_in[19];
  const float* head_W = (const float*)d_in[20];
  const float* head_b = (const float*)d_in[21];
  const float* tail_W = (const float*)d_in[22];
  const float* tail_b = (const float*)d_in[23];
  const float* self_W = (const float*)d_in[24];
  const float* self_b = (const float*)d_in[25];
  const float* e2e_W  = (const float*)d_in[26];
  const float* e2e_b  = (const float*)d_in[27];
  const float* score_W = (const float*)d_in[28];
  const float* score_b = (const float*)d_in[29];
  float* out = (float*)d_out;

  // workspace carve-out
  char* w = (char*)d_ws;
  auto alloc = [&](size_t nf) -> float* {
    float* p = (float*)w; w += align_up(nf * sizeof(float)); return p;
  };
  float* qh     = alloc((size_t)B_ * Q_ * D_);
  float* qlast  = alloc((size_t)B_ * D_);
  float* qvec   = alloc((size_t)B_ * D_);
  float* Wf     = alloc((size_t)B_ * F_);
  float* Wt     = alloc((size_t)B_ * F_);
  float* wmax   = alloc(32);
  float* e2fsm  = alloc((size_t)B_ * E_);
  float* norm   = alloc((size_t)B_ * F_);
  float* pr     = alloc((size_t)B_ * E_);
  float* pracc  = alloc((size_t)B_ * E_);
  float* lfe    = alloc((size_t)B_ * F_ * D_);     // local_fact_emb
  float* fs     = alloc((size_t)B_ * F_ * D_);     // fact self-linear -> e2f_emb in-place
  float* tl     = alloc((size_t)B_ * F_ * D_);     // tail
  float* lee0   = alloc((size_t)B_ * E_ * D_);
  float* lee1   = alloc((size_t)B_ * E_ * D_);
  float* hd     = alloc((size_t)B_ * E_ * D_);     // head, then entity-self (es)
  float* acc    = alloc((size_t)B_ * E_ * D_);     // scatter of tail
  float* nxt    = alloc((size_t)B_ * E_ * 3 * D_); // concat input to e2e

  const int MF = B_ * F_;     // 96000 fact rows
  const int ME = B_ * E_;     // 16000 entity rows
  const dim3 gF(MF / 16 / 4, (D_ + 15) / 16);   // (1500, 7)
  const dim3 gE(ME / 16 / 4, (D_ + 15) / 16);   // (250, 7)

  // pagerank_f = q2e_adj_mat[:,:,0]
  copy_kernel<<<(ME + 255) / 256, 256, 0, stream>>>(pr, q2e_adj, ME);
  // LSTM
  lstm_kernel<<<B_, 128, 0, stream>>>(word_emb, query_text, lstm_Wih, lstm_Whh,
                                      lstm_bih, lstm_bhh, qh, qlast);
  // local_fact_emb = gather(rel_emb) @ rel_W^T + rel_b   (WMMA, K=600)
  wmma_gemm_kernel<2 * WD_><<<gF, 128, 0, stream>>>(nullptr, kb_fact_rel, rel_emb,
                                                    rel_W, rel_b, lfe, MF, 0);
  // local_entity_emb = gather(entity_emb) @ ent_W^T + ent_b   (WMMA, K=300)
  wmma_gemm_kernel<WD_><<<gE, 128, 0, stream>>>(nullptr, local_entity, entity_emb,
                                                ent_W, ent_b, lee0, ME, 0);
  // attention -> Wf, W_tilde, e2f_softmax
  attn_wf_kernel<<<MF / 8, 256, 0, stream>>>(qh, lfe, query_text, Wf);
  rowmax_kernel<<<B_, 256, 0, stream>>>(Wf, wmax);
  fill_kernel<<<(ME + 255) / 256, 256, 0, stream>>>(e2fsm, ME, 0.0f);
  wtilde_kernel<<<(MF + 255) / 256, 256, 0, stream>>>(Wf, wmax, e2f_entity, Wt, e2fsm);

  float* lee = lee0;
  float* leeN = lee1;
  for (int i = 0; i < LAYERS_; ++i) {
    const float* q2eWi  = q2e_W  + (size_t)i * D_ * D_;
    const float* q2ebi  = q2e_b  + (size_t)i * D_;
    const float* headWi = head_W + (size_t)i * D_ * D_;
    const float* headbi = head_b + (size_t)i * D_;
    const float* tailWi = tail_W + (size_t)i * D_ * D_;
    const float* tailbi = tail_b + (size_t)i * D_;
    const float* selfWi = self_W + (size_t)i * D_ * D_;
    const float* selfbi = self_b + (size_t)i * D_;
    const float* e2eWi  = e2e_W  + (size_t)i * D_ * 3 * D_;
    const float* e2ebi  = e2e_b  + (size_t)i * D_;

    fill_kernel<<<(ME * D_ + 255) / 256, 256, 0, stream>>>(acc, ME * D_, 0.0f);
    fill_kernel<<<(ME + 255) / 256, 256, 0, stream>>>(pracc, ME, 0.0f);
    q2e_kernel<<<B_, 128, 0, stream>>>(qlast, q2eWi, q2ebi, qvec);
    // head = lee @ head_W^T + head_b
    wmma_gemm_kernel<D_><<<gE, 128, 0, stream>>>(lee, nullptr, nullptr,
                                                 headWi, headbi, hd, ME, 0);
    // fs = lfe @ self_W^T + self_b
    wmma_gemm_kernel<D_><<<gF, 128, 0, stream>>>(lfe, nullptr, nullptr,
                                                 selfWi, selfbi, fs, MF, 0);
    e2f_norm_kernel<<<(MF + 255) / 256, 256, 0, stream>>>(Wt, pr, e2fsm, e2f_entity, norm);
    e2f_emb_kernel<<<(MF * D_ + 255) / 256, 256, 0, stream>>>(fs, hd, e2f_entity, norm);
    // tail = e2f_emb @ tail_W^T + tail_b
    wmma_gemm_kernel<D_><<<gF, 128, 0, stream>>>(fs, nullptr, nullptr,
                                                 tailWi, tailbi, tl, MF, 0);
    scatter_tail_kernel<<<(MF * D_ + 255) / 256, 256, 0, stream>>>(tl, f2e_entity, acc);
    // es = lee @ self_W^T + self_b   (reuse hd)
    wmma_gemm_kernel<D_><<<gE, 128, 0, stream>>>(lee, nullptr, nullptr,
                                                 selfWi, selfbi, hd, ME, 0);
    scatter_pr_kernel<<<(MF + 255) / 256, 256, 0, stream>>>(norm, f2e_entity, pracc);
    pr_update_kernel<<<(ME + 255) / 256, 256, 0, stream>>>(pr, pracc);
    // nxt = concat(lee, q2e, 3*relu(es+acc)); lee_next = relu(nxt @ e2e_W^T + e2e_b)
    concat_kernel<<<(ME * D_ + 255) / 256, 256, 0, stream>>>(lee, qvec, hd, acc, nxt);
    wmma_gemm_kernel<3 * D_><<<gE, 128, 0, stream>>>(nxt, nullptr, nullptr,
                                                     e2eWi, e2ebi, leeN, ME, 1);
    float* t = lee; lee = leeN; leeN = t;
  }

  score_kernel<<<(ME + 255) / 256, 256, 0, stream>>>(lee, score_W, score_b,
                                                     local_entity, out);
}
